// multihiddenlayer_1073741824444
// MI455X (gfx1250) — matrix operand order repaired and checked by reading, not compiled, not run on hardware
//
#include <hip/hip_runtime.h>
#include <stdint.h>

typedef __attribute__((ext_vector_type(16))) _Float16 v16h;
typedef __attribute__((ext_vector_type(8)))  float    v8f;
typedef __attribute__((ext_vector_type(4)))  unsigned int u32x4;
typedef __attribute__((ext_vector_type(8)))  int      i32x8;
typedef __attribute__((ext_vector_type(4)))  int      i32x4;

#define N_ROWS 65536
#define D_IN   784
#define HID    20
#define D_OUT  10
#define N_MID  28
#define KCH    25   // ceil(784/32) K-chunks for fc1

// Low 32 bits of a flat shared-space address are the LDS byte offset
// (ISA 00_overview §10.2).
__device__ __forceinline__ unsigned lds_offset(const void* p) {
  return (unsigned)(uintptr_t)p;
}

// TDM 2D tile load: 16 rows x 32 f32 columns, row stride = 784 elements.
// tensor_dim0 = columns remaining -> hardware zero-fills OOB reads (free
// K-padding on the last chunk). Tracked by TENSORcnt.
__device__ __forceinline__ void tdm_load_tile_f32(const float* gptr,
                                                  unsigned lds_byte_addr,
                                                  unsigned cols_remaining,
                                                  unsigned row_stride_elems) {
  uint64_t ga = (uint64_t)(uintptr_t)gptr;
  u32x4 g0;
  g0[0] = 1u;                                                // count=1
  g0[1] = lds_byte_addr;                                     // lds_addr
  g0[2] = (unsigned)(ga & 0xFFFFFFFFu);                      // global_addr[31:0]
  g0[3] = (unsigned)((ga >> 32) & 0x01FFFFFFu) | (2u << 30); // addr[56:32] | type=2
  i32x8 g1;
  g1[0] = (int)(2u << 16);                                   // data_size=4B
  g1[1] = (int)((cols_remaining & 0xFFFFu) << 16);           // tensor_dim0[15:0]
  g1[2] = (int)((cols_remaining >> 16) | (16u << 16));       // dim0 hi | tensor_dim1=16
  g1[3] = (int)(32u << 16);                                  // dim1 hi | tile_dim0=32
  g1[4] = (int)16;                                           // tile_dim1=16, tile_dim2=0
  g1[5] = (int)row_stride_elems;                             // tensor_dim0_stride lo32
  g1[6] = 0;
  g1[7] = 0;
  i32x4 z4 = {0, 0, 0, 0};
#if __clang_major__ >= 23
  i32x8 z8 = {0, 0, 0, 0, 0, 0, 0, 0};
  __builtin_amdgcn_tensor_load_to_lds(g0, g1, z4, z4, z8, 0);
#else
  __builtin_amdgcn_tensor_load_to_lds(g0, g1, z4, z4, 0);
#endif
}

__device__ __forceinline__ v8f wmma_f16(v16h a, v16h b, v8f c) {
  return __builtin_amdgcn_wmma_f32_16x16x32_f16(false, a, false, b, (short)0, c,
                                                false, false);
}

__global__ __launch_bounds__(256) void mlp30_kernel(
    const float* __restrict__ X, const float* __restrict__ W1,
    const float* __restrict__ b1, const float* __restrict__ Wmid,
    const float* __restrict__ bmid, const float* __restrict__ W30,
    const float* __restrict__ b30, float* __restrict__ Y) {
  // Per-wave staging
  __shared__ float    xbuf[8][2][16][32];                    // 32 KB (TDM target)
  __shared__ _Float16 hbuf[8][16][32];                       //  8 KB activations
  // Block-wide pre-baked weight fragments (B-matrix lane layout, f16, padded)
  __shared__ __align__(16) _Float16 w1f[KCH][2][32][16];     // 50 KB
  __shared__ __align__(16) _Float16 wmf[N_MID][2][32][16];   // 56 KB
  __shared__ __align__(16) _Float16 w3f[32][16];             //  1 KB
  __shared__ float b1p[32];                                  // padded biases
  __shared__ float bmp[N_MID][32];
  __shared__ float b3p[32];

  const int tid  = (int)threadIdx.x;
  const int w    = tid >> 5;
  const int lane = tid & 31;
  const int mr   = lane & 15;               // A: row M, B/C: col N
  const int rowb = (lane < 16) ? 0 : 8;     // C/D: M base for this half-wave
  const int akb  = (lane < 16) ? 0 : 16;    // f16 A-matrix K base
  const int rowBase = (int)blockIdx.x * 128 + w * 16;

  // ---- One-time per block: bake weights into WMMA-B fragment layout --------
  // element (kc|l, nt, ln, i): N=(ln&15)+16*nt, K=32*kc + (ln<16?0:16) + i
  for (int idx = tid; idx < KCH * 1024; idx += 256) {        // 25600 = 100*256
    int i = idx & 15, ln = (idx >> 4) & 31, nt = (idx >> 9) & 1, kc = idx >> 10;
    int n = (ln & 15) + nt * 16;
    int k = kc * 32 + ((ln < 16) ? 0 : 16) + i;
    float v = (n < HID && k < D_IN) ? W1[(size_t)n * D_IN + k] : 0.0f;
    w1f[kc][nt][ln][i] = (_Float16)v;
  }
  for (int idx = tid; idx < N_MID * 1024; idx += 256) {      // 28672 = 112*256
    int i = idx & 15, ln = (idx >> 4) & 31, nt = (idx >> 9) & 1, l = idx >> 10;
    int n = (ln & 15) + nt * 16;
    int k = ((ln < 16) ? 0 : 16) + i;
    float v = (n < HID && k < HID) ? Wmid[((size_t)l * HID + n) * HID + k] : 0.0f;
    wmf[l][nt][ln][i] = (_Float16)v;
  }
  if (tid < 512) {                                           // 512 = 2*256
    int idx = tid + (tid >= 256 ? 0 : 0);
    (void)idx;
  }
  for (int idx = tid; idx < 512; idx += 256) {
    int i = idx & 15, ln = idx >> 4;
    int n = ln & 15;
    int k = ((ln < 16) ? 0 : 16) + i;
    float v = (n < D_OUT && k < HID) ? W30[n * HID + k] : 0.0f;
    w3f[ln][i] = (_Float16)v;
  }
  for (int idx = tid; idx < 32 + N_MID * 32 + 32; idx += 256) {  // 960 floats
    if (idx < 32) {
      b1p[idx] = (idx < HID) ? b1[idx] : 0.0f;
    } else if (idx < 32 + N_MID * 32) {
      int j = idx - 32, l = j >> 5, c = j & 31;
      bmp[l][c] = (c < HID) ? bmid[l * HID + c] : 0.0f;
    } else {
      int c = idx - (32 + N_MID * 32);
      b3p[c] = (c < D_OUT) ? b30[c] : 0.0f;
    }
  }
  __syncthreads();

  float*    xb0 = &xbuf[w][0][0][0];
  float*    xb1 = &xbuf[w][1][0][0];
  _Float16* hb  = &hbuf[w][0][0];

  // ---------------- fc1: h = relu(x @ W1^T + b1), K=784 in 25 chunks of 32 --
  v8f acc0, acc1;
  {
    float bv0 = b1p[mr];
    float bv1 = b1p[16 + mr];
    for (int r = 0; r < 8; ++r) { acc0[r] = bv0; acc1[r] = bv1; }
  }

  tdm_load_tile_f32(X + (size_t)rowBase * D_IN, lds_offset(xb0), D_IN, D_IN);

  for (int kc = 0; kc < KCH; ++kc) {
    const float* xb = (kc & 1) ? xb1 : xb0;
    if (kc + 1 < KCH) {   // prefetch next chunk into the other buffer
      float* nxt = (kc & 1) ? xb0 : xb1;
      tdm_load_tile_f32(X + (size_t)rowBase * D_IN + (size_t)(kc + 1) * 32,
                        lds_offset(nxt), (unsigned)(D_IN - (kc + 1) * 32), D_IN);
      __builtin_amdgcn_s_wait_tensorcnt(1);   // oldest (current chunk) done
    } else {
      __builtin_amdgcn_s_wait_tensorcnt(0);
    }
    asm volatile("" ::: "memory");

    // A fragment (16x32 f16) from the staged x tile
    v16h a;
    for (int i = 0; i < 16; ++i) {
      int k = akb + i;
      a[i] = (_Float16)xb[mr * 32 + k];
    }
    // B fragments: straight 32-byte lane-contiguous LDS reads (2x ds_load_b128)
    v16h bA = *(const v16h*)&w1f[kc][0][lane][0];
    v16h bB = *(const v16h*)&w1f[kc][1][lane][0];
    acc0 = wmma_f16(a, bA, acc0);
    acc1 = wmma_f16(a, bB, acc1);
  }

  // ReLU + pack activations to the f16 LDS tile (C layout -> A layout)
  for (int r = 0; r < 8; ++r) {
    hb[(rowb + r) * 32 + mr]      = (_Float16)fmaxf(acc0[r], 0.0f);
    hb[(rowb + r) * 32 + 16 + mr] = (_Float16)fmaxf(acc1[r], 0.0f);
  }
  asm volatile("s_wait_dscnt 0" ::: "memory");

  // ---------------- fc2..fc29: 28 x (20->20) + ReLU, padded 20->32 ----------
  for (int l = 0; l < N_MID; ++l) {
    v16h a;
    for (int i = 0; i < 16; ++i) {
      int k = akb + i;
      a[i] = hb[mr * 32 + k];
    }
    v16h bA = *(const v16h*)&wmf[l][0][lane][0];
    v16h bB = *(const v16h*)&wmf[l][1][lane][0];
    float bv0 = bmp[l][mr];
    float bv1 = bmp[l][16 + mr];
    v8f c0, c1;
    for (int r = 0; r < 8; ++r) { c0[r] = bv0; c1[r] = bv1; }
    c0 = wmma_f16(a, bA, c0);
    c1 = wmma_f16(a, bB, c1);
    // LDS is in-order per wave: reads above complete before these stores land
    for (int r = 0; r < 8; ++r) {
      hb[(rowb + r) * 32 + mr]      = (_Float16)fmaxf(c0[r], 0.0f);
      hb[(rowb + r) * 32 + 16 + mr] = (_Float16)fmaxf(c1[r], 0.0f);
    }
    asm volatile("s_wait_dscnt 0" ::: "memory");
  }

  // ---------------- fc30: 20 -> 10, no ReLU, single WMMA --------------------
  v8f co;
  {
    float bv = b3p[mr];
    for (int r = 0; r < 8; ++r) co[r] = bv;
  }
  v16h aO;
  for (int i = 0; i < 16; ++i) {
    int k = akb + i;
    aO[i] = hb[mr * 32 + k];
  }
  v16h bO = *(const v16h*)&w3f[lane][0];
  co = wmma_f16(aO, bO, co);

  if (mr < D_OUT) {
    for (int r = 0; r < 8; ++r)
      Y[(size_t)(rowBase + rowb + r) * D_OUT + mr] = co[r];
  }
}

extern "C" void kernel_launch(void* const* d_in, const int* in_sizes, int n_in,
                              void* d_out, int out_size, void* d_ws,
                              size_t ws_size, hipStream_t stream) {
  (void)in_sizes; (void)n_in; (void)out_size; (void)d_ws; (void)ws_size;
  const float* X    = (const float*)d_in[0];
  const float* W1   = (const float*)d_in[1];
  const float* b1   = (const float*)d_in[2];
  const float* Wmid = (const float*)d_in[3];
  const float* bmid = (const float*)d_in[4];
  const float* W30  = (const float*)d_in[5];
  const float* b30  = (const float*)d_in[6];
  float* Y = (float*)d_out;

  dim3 grid(N_ROWS / 128);  // 512 blocks x 8 waves x 16 rows
  dim3 block(256);
  hipLaunchKernelGGL(mlp30_kernel, grid, block, 0, stream,
                     X, W1, b1, Wmid, bmid, W30, b30, Y);
}
